// DynSMHALayer_30253749633126
// MI455X (gfx1250) — matrix operand order repaired and checked by reading, not compiled, not run on hardware
//
#include <hip/hip_runtime.h>
#include <hip/hip_bf16.h>
#include <math.h>
#include <stdint.h>

// ---- problem constants (match reference) ----
#define B_ 2
#define T_ 2048
#define H_ 1024
#define D_ 64
#define M_ 16
#define NT_ (T_ / 32)   // key tiles of 32

typedef __attribute__((ext_vector_type(16))) __bf16 v16bf;
typedef __attribute__((ext_vector_type(8)))  float  v8f;

// ---------------------------------------------------------------------------
// WMMA bf16 16x16x32 fragment index helpers (CDNA5 ISA 7.12.2):
//  A (16x32, MxK): lane = M%16 + 16*((K%16)>=8), elem = (K%8) + (K>=16 ? 8:0)
//  B (32x16, KxN): lane = N%16 + 16*((K%16)>=8), elem = (K%8) + (K>=16 ? 8:0)
//  C (16x16, MxN): row  = vgpr + 8*(lane/16),    col  = lane%16
// ---------------------------------------------------------------------------
__device__ __forceinline__ int a_lane(int row, int k) {
  return (row & 15) + (((k & 15) >= 8) ? 16 : 0);
}
__device__ __forceinline__ int a_elem(int k) {
  return (k & 7) + ((k >= 16) ? 8 : 0);
}
__device__ __forceinline__ int b_lane(int k, int n) {
  return (n & 15) + (((k & 15) >= 8) ? 16 : 0);
}
__device__ __forceinline__ int b_elem(int k) { return (k & 7) + ((k >= 16) ? 8 : 0); }

__device__ __forceinline__ v8f wmma_bf16(v16bf a, v16bf b, v8f c) {
  return __builtin_amdgcn_wmma_f32_16x16x32_bf16(false, a, false, b, (short)0, c,
                                                 false, false);
}

// ---------------------------------------------------------------------------
// Async global->LDS (CDNA5 GLOBAL_LOAD_ASYNC_TO_LDS_B128, ASYNCcnt-tracked).
// Probe round 2 showed the builtin exists and expects typed pointers:
//   param1 = 'vector_size(16) int' in addrspace(1)  (global source)
//   param2 = same vector type in addrspace(3)       (LDS destination)
// Generic->AS casts go through integers: flat-LDS low 32 bits == LDS offset.
// ---------------------------------------------------------------------------
#if __has_builtin(__builtin_amdgcn_global_load_async_to_lds_b128) && \
    __has_builtin(__builtin_amdgcn_s_wait_asynccnt)
#define USE_ASYNC 1
#else
#define USE_ASYNC 0
#endif

typedef int vsi4 __attribute__((vector_size(16)));
typedef __attribute__((address_space(1))) vsi4 as1_vsi4;
typedef __attribute__((address_space(3))) vsi4 as3_vsi4;

__device__ __forceinline__ void copy16_async(const void* g, void* l) {
#if USE_ASYNC
  __builtin_amdgcn_global_load_async_to_lds_b128(
      (as1_vsi4*)(uintptr_t)g, (as3_vsi4*)(uint32_t)(uintptr_t)l, 0, 0);
#else
  *(uint4*)l = *(const uint4*)g;
#endif
}
__device__ __forceinline__ void wait_async_all() {
#if USE_ASYNC
  __builtin_amdgcn_s_wait_asynccnt(0);
#endif
}

// ---------------------------------------------------------------------------
// Kernel 0: zero the entropy accumulators
// ---------------------------------------------------------------------------
__global__ void init_kernel(float* entsum) {
  if (threadIdx.x < B_ * M_) entsum[threadIdx.x] = 0.0f;
}

// ---------------------------------------------------------------------------
// Kernel 1: QKV projections.  [T,H] x [H,D] per (b,m), bf16 WMMA, fp32 accum.
// Writes Q/K/V to workspace ALREADY IN WMMA FRAGMENT LAYOUT so the attention
// kernel can stage tiles with contiguous async b128 copies (no swizzle):
//   qfrag[bm][rowtile(T/16)][kc(2)][lane(32)][elem(16)]             (A-frags)
//   kv   [bm][jt(T/32)] = { K: [kc2][keyhalf2][32][16] (4KB),
//                           V: [dt4][32][16]          (4KB) }  8KB/tile
// grid = (B*M, T/64), 256 threads (8 waves)
// ---------------------------------------------------------------------------
__global__ __launch_bounds__(256) void qkv_kernel(
    const float* __restrict__ X, const float* __restrict__ Wq,
    const float* __restrict__ Wk, const float* __restrict__ Wv,
    __bf16* __restrict__ qfrag, __bf16* __restrict__ kv) {
  __shared__ alignas(32) __bf16 lA[4][32][16];      // X frags
  __shared__ alignas(32) __bf16 lB[3][4][32][16];   // W frags
  const int bm = blockIdx.x, b = bm / M_, m = bm % M_;
  const int t0 = blockIdx.y * 64;
  const int tid = threadIdx.x, w = tid >> 5, lane = tid & 31;
  const int rt = w & 3, cp = w >> 2;
  const float* Ws[3] = {Wq, Wk, Wv};
  v8f acc[3][2] = {};
  const int arow = tid >> 2, akb = (tid & 3) * 8;
  const int brow = tid >> 3, bnb = (tid & 7) * 8;

  for (int kk = 0; kk < H_; kk += 32) {
    __syncthreads();
    {  // stage X tile -> A fragments (fp32 -> bf16)
      const float4* src =
          (const float4*)&X[(size_t)(b * T_ + t0 + arow) * H_ + kk + akb];
      float4 f0 = src[0], f1 = src[1];
      float vals[8] = {f0.x, f0.y, f0.z, f0.w, f1.x, f1.y, f1.z, f1.w};
#pragma unroll
      for (int i = 0; i < 8; ++i) {
        int kc = akb + i;
        lA[arow >> 4][a_lane(arow, kc)][a_elem(kc)] = (__bf16)vals[i];
      }
      if (kk + 32 < H_)
        __builtin_prefetch(&X[(size_t)(b * T_ + t0 + arow) * H_ + kk + 32 + akb],
                           0, 1);
    }
#pragma unroll
    for (int j = 0; j < 3; ++j) {  // stage Wq/Wk/Wv tiles -> B fragments
      const float4* src =
          (const float4*)&Ws[j][((size_t)m * H_ + kk + brow) * D_ + bnb];
      float4 f0 = src[0], f1 = src[1];
      float vals[8] = {f0.x, f0.y, f0.z, f0.w, f1.x, f1.y, f1.z, f1.w};
#pragma unroll
      for (int i = 0; i < 8; ++i) {
        int n = bnb + i;
        lB[j][n >> 4][b_lane(brow, n)][b_elem(brow)] = (__bf16)vals[i];
      }
    }
    __syncthreads();
    v16bf af = *(const v16bf*)&lA[rt][lane][0];
#pragma unroll
    for (int j = 0; j < 3; ++j)
#pragma unroll
      for (int c = 0; c < 2; ++c) {
        v16bf bf = *(const v16bf*)&lB[j][2 * cp + c][lane][0];
        acc[j][c] = wmma_bf16(af, bf, acc[j][c]);
      }
  }

  // ---- store C tiles directly into fragment layouts ----
  const size_t qbase = (size_t)bm * (T_ / 16) * 1024;
  const size_t kvbase = (size_t)bm * NT_ * 4096;
  const int l15 = lane & 15, lh = lane >> 4;
#pragma unroll
  for (int c = 0; c < 2; ++c)
#pragma unroll
    for (int i = 0; i < 8; ++i) {
      int row = t0 + 16 * rt + i + 8 * lh;   // global t index
      int row16 = i + 8 * lh;                // row within 16-tile
      // Q as A-fragment: chunk kc = cp, dd = c*16 + l15
      {
        int lq = row16 + 16 * ((l15 >= 8) ? 1 : 0);
        int eq = (l15 & 7) + (c ? 8 : 0);
        qfrag[qbase + ((size_t)(row >> 4) * 2 + cp) * 512 + lq * 16 + eq] =
            (__bf16)acc[0][c][i];
      }
      int jt = row >> 5, keyloc = row & 31;
      // K as B-fragment of Q*K^T: kdim = dd = c*16+l15, n = key
      {
        int lk = (keyloc & 15) + 16 * ((l15 >= 8) ? 1 : 0);
        kv[kvbase + (size_t)jt * 4096 +
           (((cp * 2) + (keyloc >> 4)) * 32 + lk) * 16 + (l15 & 7) + (c ? 8 : 0)] =
            (__bf16)acc[1][c][i];
      }
      // V as B-fragment of P*V: kdim = keyloc, n = d%16, dt = 2*cp+c
      {
        int lv = l15 + 16 * (((keyloc & 15) >= 8) ? 1 : 0);
        kv[kvbase + (size_t)jt * 4096 + 2048 +
           ((2 * cp + c) * 32 + lv) * 16 + (keyloc & 7) + ((keyloc >> 4) ? 8 : 0)] =
            (__bf16)acc[2][c][i];
      }
    }
}

// ---------------------------------------------------------------------------
// Kernel 2: flash attention + entropy.  Each wave owns 16 q-rows, block covers
// 128 q-rows, streams 32-key tiles via double-buffered ASYNC global->LDS
// copies of the pre-swizzled K/V fragment blocks.
// entropy_row = (m + log l) - t/l  with l = sum e^{s-m}, t = sum s*e^{s-m}.
// grid = (B*M, T/128)
// ---------------------------------------------------------------------------
__global__ __launch_bounds__(256) void attn_kernel(
    const __bf16* __restrict__ qfrag, const __bf16* __restrict__ kv,
    float* __restrict__ sha, float* __restrict__ entsum) {
  __shared__ alignas(64) __bf16 lKV[2][4096];     // double-buffered 8KB tile
  __shared__ alignas(32) __bf16 lP[8][32][16];    // per-wave P in A-layout
  const int bm = blockIdx.x, b = bm / M_, m = bm % M_;
  const int tid = threadIdx.x, w = tid >> 5, lane = tid & 31;
  const int q0 = blockIdx.y * 128 + w * 16;

  // Q fragments: contiguous b128 loads from the fragment-layout workspace
  v16bf aq[2];
  {
    const __bf16* qf =
        qfrag + (size_t)bm * (T_ / 16) * 1024 + (size_t)(q0 >> 4) * 1024;
    aq[0] = *(const v16bf*)&qf[(0 * 32 + lane) * 16];
    aq[1] = *(const v16bf*)&qf[(1 * 32 + lane) * 16];
  }

  float rm[8], rl[8], rt_[8];
#pragma unroll
  for (int i = 0; i < 8; ++i) { rm[i] = -3.0e38f; rl[i] = 0.f; rt_[i] = 0.f; }
  v8f oacc[4] = {};

  const char* kvsrc = (const char*)kv + (size_t)bm * NT_ * 8192;
  // issue tile 0 into buffer 0 (32 bytes per thread, ASYNCcnt-tracked)
  copy16_async(kvsrc + (size_t)tid * 32, (char*)&lKV[0][0] + tid * 32);
  copy16_async(kvsrc + (size_t)tid * 32 + 16, (char*)&lKV[0][0] + tid * 32 + 16);

  for (int jt = 0; jt < NT_; ++jt) {
    const int buf = jt & 1;
    wait_async_all();
    __syncthreads();
    if (jt + 1 < NT_) {  // prefetch next tile into the other buffer
      const char* src = kvsrc + (size_t)(jt + 1) * 8192 + (size_t)tid * 32;
      char* dst = (char*)&lKV[buf ^ 1][0] + tid * 32;
      copy16_async(src, dst);
      copy16_async(src + 16, dst + 16);
    }

    // S tile 16x32 = two 16x16 accumulators, K-dim = D = 2x32
    v8f s0 = {}, s1 = {};
#pragma unroll
    for (int kc = 0; kc < 2; ++kc) {
      v16bf bk0 = *(const v16bf*)&lKV[buf][((kc * 2 + 0) * 32 + lane) * 16];
      v16bf bk1 = *(const v16bf*)&lKV[buf][((kc * 2 + 1) * 32 + lane) * 16];
      s0 = wmma_bf16(aq[kc], bk0, s0);
      s1 = wmma_bf16(aq[kc], bk1, s1);
    }

    float al[8];
#pragma unroll
    for (int i = 0; i < 8; ++i) {
      float a0 = s0[i] * 0.125f, a1 = s1[i] * 0.125f;  // 1/sqrt(D)
      float mx = fmaxf(a0, a1);
#pragma unroll
      for (int off = 8; off > 0; off >>= 1)
        mx = fmaxf(mx, __shfl_xor(mx, off, 32));
      float mnew = fmaxf(rm[i], mx);
      float p0 = __expf(a0 - mnew), p1 = __expf(a1 - mnew);
      float rs = p0 + p1, rss = p0 * a0 + p1 * a1;
#pragma unroll
      for (int off = 8; off > 0; off >>= 1) {
        rs += __shfl_xor(rs, off, 32);
        rss += __shfl_xor(rss, off, 32);
      }
      float alpha = __expf(rm[i] - mnew);
      rl[i] = rl[i] * alpha + rs;
      rt_[i] = rt_[i] * alpha + rss;
      rm[i] = mnew;
      al[i] = alpha;
      s0[i] = p0;
      s1[i] = p1;
    }
#pragma unroll
    for (int dt = 0; dt < 4; ++dt)
#pragma unroll
      for (int i = 0; i < 8; ++i) oacc[dt][i] *= al[i];

    {  // P (C-layout) -> per-wave LDS scratch in A-layout (keys = K dim)
      int lh = lane >> 4, lcol = lane & 15;
      int alb = (lcol >= 8) ? 16 : 0;
      int ae0 = lane & 7;
#pragma unroll
      for (int i = 0; i < 8; ++i) {
        int arow = i + 8 * lh;
        lP[w][alb + arow][ae0]     = (__bf16)s0[i];  // keys 0..15
        lP[w][alb + arow][ae0 + 8] = (__bf16)s1[i];  // keys 16..31
      }
    }
    v16bf pa = *(const v16bf*)&lP[w][lane][0];
#pragma unroll
    for (int dt = 0; dt < 4; ++dt) {
      v16bf bv = *(const v16bf*)&lKV[buf][2048 + (dt * 32 + lane) * 16];
      oacc[dt] = wmma_bf16(pa, bv, oacc[dt]);
    }
  }

  // epilogue: normalize, write all_sha_outputs [B,T,M,D] (fp32)
#pragma unroll
  for (int i = 0; i < 8; ++i) {
    float inv = 1.0f / rl[i];
    int row = q0 + i + 8 * (lane >> 4);
    size_t o = (((size_t)(b * T_ + row)) * M_ + m) * (size_t)D_ + (lane & 15);
#pragma unroll
    for (int dt = 0; dt < 4; ++dt) sha[o + dt * 16] = oacc[dt][i] * inv;
  }
  if ((lane & 15) == 0) {  // lanes 0 & 16 own disjoint rows
    float es = 0.f;
#pragma unroll
    for (int i = 0; i < 8; ++i)
      es += (rm[i] + __logf(rl[i])) - rt_[i] / rl[i];
    atomicAdd(&entsum[b * M_ + m], es);
  }
}

// ---------------------------------------------------------------------------
// Kernel 3: gating (tiny; single thread handles B=2 x M=16 serially).
// ---------------------------------------------------------------------------
__global__ void gate_kernel(const float* __restrict__ gates,
                            const float* __restrict__ entsum,
                            float* __restrict__ out_pre,
                            float* __restrict__ out_mask,
                            float* __restrict__ out_fb,
                            float* __restrict__ nm) {
  if (threadIdx.x != 0 || blockIdx.x != 0) return;
  int fb_total = 0;
  for (int b = 0; b < B_; ++b) {
    float aff[M_];
    for (int m = 0; m < M_; ++m) aff[m] = -entsum[b * M_ + m] / (float)T_;
    float mu = 0.f;
    for (int m = 0; m < M_; ++m) mu += aff[m];
    mu /= (float)M_;
    float var = 0.f;
    for (int m = 0; m < M_; ++m) var += (aff[m] - mu) * (aff[m] - mu);
    float sd = sqrtf(var / (float)(M_ - 1));  // ddof=1
    for (int m = 0; m < M_; ++m) aff[m] = (aff[m] - mu) / (sd + 1e-9f);
    float pre[M_], mask[M_];
    int nact = 0;
    for (int m = 0; m < M_; ++m) {
      pre[m] = aff[m] - 1.0f / (1.0f + __expf(-gates[m]));
      mask[m] = (pre[m] > 0.f) ? 1.f : 0.f;
      nact += (pre[m] > 0.f) ? 1 : 0;
    }
    if (nact == 0) {  // fallback: top-2 by affinity
      ++fb_total;
      int i1 = 0;
      for (int m = 1; m < M_; ++m) if (aff[m] > aff[i1]) i1 = m;
      int i2 = (i1 == 0) ? 1 : 0;
      for (int m = 0; m < M_; ++m)
        if (m != i1 && aff[m] > aff[i2]) i2 = m;
      mask[i1] = 1.f;
      mask[i2] = 1.f;
    }
    float na = 0.f;
    for (int m = 0; m < M_; ++m) na += mask[m];
    float denom = fmaxf(na, 1.f);
    for (int m = 0; m < M_; ++m) {
      out_pre[b * M_ + m] = pre[m];
      out_mask[b * M_ + m] = mask[m];
      nm[b * M_ + m] = mask[m] / denom;
    }
  }
  out_fb[0] = (float)fb_total;
}

// ---------------------------------------------------------------------------
// Kernel 4a: dynamic_o_proj = sum_m o_weights[m]*nm[b,m], written directly in
// B-fragment layout: opfrag[b][hblk(H/128)][kc2][ht8][lane32][elem16]
// ---------------------------------------------------------------------------
__global__ __launch_bounds__(256) void oproj_kernel(
    const float* __restrict__ ow, const float* __restrict__ nm,
    __bf16* __restrict__ opfrag) {
  int idx = blockIdx.x * 256 + threadIdx.x;
  if (idx >= B_ * D_ * H_) return;
  int b = idx / (D_ * H_);
  int dh = idx % (D_ * H_);
  int d = dh / H_, h = dh % H_;
  float s = 0.f;
#pragma unroll
  for (int m = 0; m < M_; ++m)
    s += ow[(size_t)m * D_ * H_ + dh] * nm[b * M_ + m];
  int dd = d & 31;
  size_t o = ((((size_t)b * (H_ / 128) + (h >> 7)) * 2 + (d >> 5)) * 8 +
              ((h & 127) >> 4)) * 512 +
             (size_t)b_lane(dd, h) * 16 + b_elem(dd);
  opfrag[o] = (__bf16)s;
}

// ---------------------------------------------------------------------------
// Kernel 4b: final = (sum_m sha*nm) @ o_proj.  Tile 32 T-rows x 128 H-cols.
// o_proj tile staged with contiguous async b128 copies (pre-swizzled).
// grid = B*(T/32)*(H/128)
// ---------------------------------------------------------------------------
__global__ __launch_bounds__(256) void final_kernel(
    const float* __restrict__ sha, const __bf16* __restrict__ opfrag,
    const float* __restrict__ nm, float* __restrict__ out) {
  __shared__ alignas(32) __bf16 lA[2][2][32][16];  // [rowtile][kc][lane][elem]
  __shared__ alignas(64) __bf16 lB2[2][8][32][16]; // [kc][ht][lane][elem] 16KB
  const int nh = H_ / 128;
  const int bx = blockIdx.x;
  const int b = bx / ((T_ / 32) * nh);
  const int r2 = bx % ((T_ / 32) * nh);
  const int t0 = (r2 / nh) * 32;
  const int h0 = (r2 % nh) * 128;
  const int tid = threadIdx.x, w = tid >> 5, lane = tid & 31;

  {  // async-stage o_proj fragments (16KB contiguous)
    const char* src = (const char*)opfrag +
                      ((size_t)b * nh + (h0 >> 7)) * 16384 + (size_t)tid * 64;
    char* dst = (char*)&lB2[0][0][0][0] + tid * 64;
#pragma unroll
    for (int c = 0; c < 4; ++c) copy16_async(src + c * 16, dst + c * 16);
  }
  float nmv[M_];
#pragma unroll
  for (int m = 0; m < M_; ++m) nmv[m] = nm[b * M_ + m];
  {  // stage combined_heads tile (32 x 64): masked expert sum -> A frags
    int row = tid >> 3, db = (tid & 7) * 8;
#pragma unroll
    for (int i = 0; i < 8; ++i) {
      int d = db + i;
      float s = 0.f;
      size_t o = ((size_t)(b * T_ + t0 + row)) * (size_t)(M_ * D_) + d;
#pragma unroll
      for (int m = 0; m < M_; ++m) s += sha[o + (size_t)m * D_] * nmv[m];
      int dd = d & 31;
      lA[row >> 4][d >> 5][a_lane(row, dd)][a_elem(dd)] = (__bf16)s;
    }
  }
  wait_async_all();
  __syncthreads();

  const int rt2 = w & 1, hp = w >> 1;
  v8f acc[2] = {};
#pragma unroll
  for (int kc = 0; kc < 2; ++kc) {
    v16bf af = *(const v16bf*)&lA[rt2][kc][lane][0];
#pragma unroll
    for (int c = 0; c < 2; ++c) {
      v16bf bf = *(const v16bf*)&lB2[kc][2 * hp + c][lane][0];
      acc[c] = wmma_bf16(af, bf, acc[c]);
    }
  }
#pragma unroll
  for (int c = 0; c < 2; ++c)
#pragma unroll
    for (int i = 0; i < 8; ++i) {
      int row = t0 + 16 * rt2 + i + 8 * (lane >> 4);
      int col = h0 + (2 * hp + c) * 16 + (lane & 15);
      out[((size_t)(b * T_ + row)) * H_ + col] = acc[c][i];
    }
}

// ---------------------------------------------------------------------------
// kernel_launch: workspace ~24.3 MB (bf16 Q frags 8MB + K/V frag tiles 16MB
// + accumulators + o_proj frags 256KB)
// ---------------------------------------------------------------------------
extern "C" void kernel_launch(void* const* d_in, const int* in_sizes, int n_in,
                              void* d_out, int out_size, void* d_ws,
                              size_t ws_size, hipStream_t stream) {
  (void)in_sizes; (void)n_in; (void)out_size; (void)ws_size;
  const float* X  = (const float*)d_in[0];
  const float* Wq = (const float*)d_in[1];
  const float* Wk = (const float*)d_in[2];
  const float* Wv = (const float*)d_in[3];
  const float* Ow = (const float*)d_in[4];
  const float* gates = (const float*)d_in[5];
  float* out = (float*)d_out;

  const size_t NQ = (size_t)B_ * M_ * (T_ / 16) * 1024;  // q frag elems (8MB)
  const size_t NKV = (size_t)B_ * M_ * NT_ * 4096;       // kv frag elems (16MB)
  char* ws = (char*)d_ws;
  __bf16* qfrag = (__bf16*)ws;
  __bf16* kv = (__bf16*)(ws + NQ * 2);
  float* entsum = (float*)(ws + NQ * 2 + NKV * 2);
  float* nm = entsum + 64;
  __bf16* opfrag = (__bf16*)(nm + 64);

  float* out_final = out;                                   // [B,T,H]
  float* out_sha   = out + (size_t)B_ * T_ * H_;            // [B,T,M,D]
  float* out_pre   = out_sha + (size_t)B_ * T_ * M_ * D_;   // [B,M]
  float* out_mask  = out_pre + B_ * M_;                     // [B,M]
  float* out_fb    = out_mask + B_ * M_;                    // [1]

  hipLaunchKernelGGL(init_kernel, dim3(1), dim3(64), 0, stream, entsum);
  hipLaunchKernelGGL(qkv_kernel, dim3(B_ * M_, T_ / 64), dim3(256), 0, stream,
                     X, Wq, Wk, Wv, qfrag, kv);
  hipLaunchKernelGGL(attn_kernel, dim3(B_ * M_, T_ / 128), dim3(256), 0, stream,
                     qfrag, kv, out_sha, entsum);
  hipLaunchKernelGGL(gate_kernel, dim3(1), dim3(32), 0, stream, gates, entsum,
                     out_pre, out_mask, out_fb, nm);
  hipLaunchKernelGGL(oproj_kernel, dim3((B_ * D_ * H_ + 255) / 256), dim3(256),
                     0, stream, Ow, nm, opfrag);
  hipLaunchKernelGGL(final_kernel, dim3(B_ * (T_ / 32) * (H_ / 128)), dim3(256),
                     0, stream, out_sha, opfrag, nm, out_final);
}
